// DGCNNCls_84920093376685
// MI455X (gfx1250) — matrix operand order repaired and checked by reading, not compiled, not run on hardware
//
#include <hip/hip_runtime.h>

typedef __attribute__((ext_vector_type(16))) _Float16 v16h;
typedef __attribute__((ext_vector_type(8)))  _Float16 v8h;
typedef __attribute__((ext_vector_type(8)))  float    v8f;

#define NEG_INF (-3.0e38f)
#define BATCH 8
#define NPTS  1024
#define KNN_K 20

// ---------------------------------------------------------------------------
// Weight f32 [O][Cin] -> padded f16  Wt[O][CP]  (zeros for c>=Cin).
// K-contiguous layout: WMMA B-fragment (N=lane%16, K=elem in table order)
// is two 8-f16 (16B) runs per lane -> 2x global_load_b128 per fragment.
// ---------------------------------------------------------------------------
__global__ void convert_wt_kernel(const float* __restrict__ W, _Float16* __restrict__ Wt,
                                  int O, int Cin, int CP) {
    int t = blockIdx.x * blockDim.x + threadIdx.x;
    if (t >= O * CP) return;
    int o = t / CP, c = t % CP;
    Wt[t] = (_Float16)(c < Cin ? W[(size_t)o * Cin + c] : 0.0f);
}

// ---------------------------------------------------------------------------
// KNN: one block per (b,n). Shifted negative squared distance to all m, then
// 20 argmax iterations (ordering matches reference up to the -xx_n constant).
// ---------------------------------------------------------------------------
template<int C>
__global__ void knn_kernel(const float* __restrict__ x, int* __restrict__ idx) {
    int bn = blockIdx.x;
    int b = bn / NPTS, n = bn % NPTS;
    const float* xb = x + (size_t)b * NPTS * C;

    __shared__ float ctr[C];
    __shared__ float dist[NPTS];
    __shared__ float sval[256];
    __shared__ int   sidx[256];

    int tid = threadIdx.x;
    for (int c = tid; c < C; c += blockDim.x) ctr[c] = xb[(size_t)n * C + c];
    __syncthreads();

    for (int m = tid; m < NPTS; m += blockDim.x) {
        const float* xm = xb + (size_t)m * C;
        float inner = 0.0f, xxm = 0.0f;
        if constexpr ((C % 4) == 0) {
            const float4* xm4 = (const float4*)xm;
#pragma unroll 4
            for (int c4 = 0; c4 < C / 4; ++c4) {
                float4 v = xm4[c4];
                inner += v.x * ctr[c4*4+0] + v.y * ctr[c4*4+1] + v.z * ctr[c4*4+2] + v.w * ctr[c4*4+3];
                xxm   += v.x * v.x + v.y * v.y + v.z * v.z + v.w * v.w;
            }
        } else {
            for (int c = 0; c < C; ++c) { float v = xm[c]; inner += v * ctr[c]; xxm += v * v; }
        }
        dist[m] = 2.0f * inner - xxm;
    }
    __syncthreads();

    for (int kk = 0; kk < KNN_K; ++kk) {
        float best = NEG_INF; int bi = 0;
        for (int m = tid; m < NPTS; m += blockDim.x)
            if (dist[m] > best) { best = dist[m]; bi = m; }
        sval[tid] = best; sidx[tid] = bi;
        __syncthreads();
        for (int s = 128; s > 0; s >>= 1) {
            if (tid < s && sval[tid + s] > sval[tid]) { sval[tid] = sval[tid + s]; sidx[tid] = sidx[tid + s]; }
            __syncthreads();
        }
        if (tid == 0) { idx[(size_t)bn * KNN_K + kk] = sidx[0]; dist[sidx[0]] = NEG_INF; }
        __syncthreads();
    }
}

// ---------------------------------------------------------------------------
// EdgeConv via WMMA: one block (8 waves) per point (b,n).
// Edge features are written to LDS *already in A-fragment order*:
//   fragment f = mtile*KC + kc ; within it lane L, elem e ->
//   row k = mtile*16 + L%16 ; col c = kc*32 + (e/8)*16 + (L/16)*8 + e%8
// so each lane's A fragment is a contiguous 32B run (2x ds_load_b128).
// B fragments come from the K-contiguous Wt[O][CP] as two v8h loads per lane.
// ---------------------------------------------------------------------------
template<int C, int O>
__global__ void edgeconv_kernel(const float* __restrict__ x, const int* __restrict__ idx,
                                const _Float16* __restrict__ Wt, const float* __restrict__ bnp,
                                float* __restrict__ out, _Float16* __restrict__ cat, int cat_off) {
    constexpr int CP = ((2 * C + 31) / 32) * 32;
    constexpr int KC = CP / 32;
    constexpr int OT = O / 16;
    constexpr int NFRAG = 2 * KC;

    int bn = blockIdx.x;
    int b = bn / NPTS, n = bn % NPTS;
    const float* xb = x + (size_t)b * NPTS * C;

    __shared__ __align__(32) _Float16 featA[NFRAG * 512];
    __shared__ int nb[KNN_K];

    int tid = threadIdx.x;
    if (tid < KNN_K) nb[tid] = idx[(size_t)bn * KNN_K + tid];
    __syncthreads();

    for (int t = tid; t < NFRAG * 512; t += blockDim.x) {
        int f = t >> 9, r = t & 511;
        int L = r >> 4, e = r & 15;
        int mtile = f / KC, kc = f % KC;
        int k = mtile * 16 + (L & 15);
        int c = kc * 32 + ((e >> 3) << 4) + ((L >> 4) << 3) + (e & 7);
        float v = 0.0f;
        if (k < KNN_K && c < 2 * C) {
            if (c < C) v = xb[(size_t)nb[k] * C + c] - xb[(size_t)n * C + c];
            else       v = xb[(size_t)n * C + (c - C)];
        }
        featA[t] = (_Float16)v;
    }
    __syncthreads();

    int wave = tid >> 5, lane = tid & 31;
    int half = lane >> 4;
    const v16h* fa = (const v16h*)featA;                 // fragment rows: 32 v16h per fragment
    const v16h* wt = (const v16h*)Wt;                    // Wt[o][c]: row o has CP/16 v16h

    for (int ot = wave; ot < OT; ot += 8) {
        v8f acc0 = {}; v8f acc1 = {};
        for (int kc = 0; kc < KC; ++kc) {
            v16h a0 = fa[kc * 32 + lane];                // M-tile 0 fragment
            v16h a1 = fa[(KC + kc) * 32 + lane];         // M-tile 1 fragment
            const _Float16* wrow = Wt + (size_t)(ot * 16 + (lane & 15)) * CP + kc * 32 + half * 8;
            v16h bf = __builtin_shufflevector(*(const v8h*)wrow, *(const v8h*)(wrow + 16), 0,1,2,3,4,5,6,7,8,9,10,11,12,13,14,15);
            acc0 = __builtin_amdgcn_wmma_f32_16x16x32_f16(false, a0, false, bf, (short)0, acc0, false, false);
            acc1 = __builtin_amdgcn_wmma_f32_16x16x32_f16(false, a1, false, bf, (short)0, acc1, false, false);
        }
        // D layout: N = lane%16, M = r + 8*(lane/16)
        int o = ot * 16 + (lane & 15);
        float ga = bnp[o], be = bnp[O + o], mu = bnp[2 * O + o], va = bnp[3 * O + o];
        float scale = ga * rsqrtf(va + 1e-5f);
        float mx = NEG_INF;
#pragma unroll
        for (int r = 0; r < 8; ++r) {
            float y0 = scale * (acc0[r] - mu) + be; y0 = y0 > 0.0f ? y0 : 0.2f * y0;
            mx = fmaxf(mx, y0);                           // tile0 edges 0..15: always valid
            int m1 = 16 + r + 8 * half;                   // tile1 edges: valid only < 20
            float y1 = scale * (acc1[r] - mu) + be; y1 = y1 > 0.0f ? y1 : 0.2f * y1;
            if (m1 < KNN_K) mx = fmaxf(mx, y1);
        }
        mx = fmaxf(mx, __shfl_xor(mx, 16, 32));
        if (lane < 16) {
            out[(size_t)bn * O + o] = mx;
            cat[(size_t)bn * 512 + cat_off + o] = (_Float16)mx;
        }
    }
}

// ---------------------------------------------------------------------------
// y = lrelu(bn5(cat @ W5^T)) : GEMM M=8192, K=512, N=512. One wave per 16x16
// tile; A fragment = two contiguous b128 loads + shuffle, B likewise two b128 loads + shuffle.
// ---------------------------------------------------------------------------
__global__ void gemm_bn_kernel(const _Float16* __restrict__ A, const _Float16* __restrict__ Wt,
                               const float* __restrict__ bnp, float* __restrict__ y) {
    int gw = (blockIdx.x * blockDim.x + threadIdx.x) >> 5;
    int lane = threadIdx.x & 31;
    const int NT = 512 / 16;
    int mt = gw / NT, nt = gw % NT;
    int Mloc = lane & 15, half = lane >> 4;

    const _Float16* arow = A + (size_t)(mt * 16 + Mloc) * 512;
    const v16h* wt = (const v16h*)Wt;

    v8f acc = {};
    for (int kc = 0; kc < 512 / 32; ++kc) {
        v8h alo = *(const v8h*)(arow + kc * 32 + half * 8);
        v8h ahi = *(const v8h*)(arow + kc * 32 + 16 + half * 8);
        v16h a = __builtin_shufflevector(alo, ahi, 0,1,2,3,4,5,6,7,8,9,10,11,12,13,14,15);
        const _Float16* wrow = Wt + (size_t)(nt * 16 + Mloc) * 512 + kc * 32 + half * 8;
        v16h bf = __builtin_shufflevector(*(const v8h*)wrow, *(const v8h*)(wrow + 16), 0,1,2,3,4,5,6,7,8,9,10,11,12,13,14,15);
        if (kc + 1 < 16) __builtin_prefetch(arow + (kc + 1) * 32, 0, 1);
        acc = __builtin_amdgcn_wmma_f32_16x16x32_f16(false, a, false, bf, (short)0, acc, false, false);
    }
    int o = nt * 16 + (lane & 15);
    float ga = bnp[o], be = bnp[512 + o], mu = bnp[1024 + o], va = bnp[1536 + o];
    float scale = ga * rsqrtf(va + 1e-5f);
#pragma unroll
    for (int r = 0; r < 8; ++r) {
        int m = mt * 16 + r + 8 * half;
        float v = scale * (acc[r] - mu) + be;
        v = v > 0.0f ? v : 0.2f * v;
        y[(size_t)m * 512 + o] = v;
    }
}

// ---------------------------------------------------------------------------
// Global max+mean pool over N: one wave per (b,o).
// ---------------------------------------------------------------------------
__global__ void pool_kernel(const float* __restrict__ y, float* __restrict__ g) {
    int gw = (blockIdx.x * blockDim.x + threadIdx.x) >> 5;
    int lane = threadIdx.x & 31;
    int b = gw / 512, o = gw % 512;
    const float* yb = y + (size_t)b * NPTS * 512 + o;
    float mx = NEG_INF, sm = 0.0f;
    for (int n = lane; n < NPTS; n += 32) {
        float v = yb[(size_t)n * 512];
        mx = fmaxf(mx, v); sm += v;
    }
    for (int s = 16; s > 0; s >>= 1) {
        mx = fmaxf(mx, __shfl_xor(mx, s, 32));
        sm += __shfl_xor(sm, s, 32);
    }
    if (lane == 0) {
        g[(size_t)b * 1024 + o]       = mx;
        g[(size_t)b * 1024 + 512 + o] = sm * (1.0f / (float)NPTS);
    }
}

// ---------------------------------------------------------------------------
// FC head: tiny (B=8 rows) — VALU dot per output element.
// ---------------------------------------------------------------------------
template<bool HAS_BN, bool LRELU>
__global__ void fc_kernel(const float* __restrict__ in, const float* __restrict__ W,
                          const float* __restrict__ bias, const float* __restrict__ bnp,
                          float* __restrict__ out, int B, int I, int O) {
    int t = blockIdx.x * blockDim.x + threadIdx.x;
    if (t >= B * O) return;
    int b = t / O, o = t % O;
    const float* ib = in + (size_t)b * I;
    const float* wo = W + (size_t)o * I;
    float s = bias[o];
    for (int i = 0; i < I; ++i) s += ib[i] * wo[i];
    if constexpr (HAS_BN) {
        float ga = bnp[o], be = bnp[O + o], mu = bnp[2 * O + o], va = bnp[3 * O + o];
        s = ga * (s - mu) * rsqrtf(va + 1e-5f) + be;
    }
    if constexpr (LRELU) s = s > 0.0f ? s : 0.2f * s;
    out[t] = s;
}

// ---------------------------------------------------------------------------
extern "C" void kernel_launch(void* const* d_in, const int* in_sizes, int n_in,
                              void* d_out, int out_size, void* d_ws, size_t ws_size,
                              hipStream_t stream) {
    const float* x   = (const float*)d_in[0];
    const float* W1  = (const float*)d_in[1];  const float* bn1 = (const float*)d_in[2];
    const float* W2  = (const float*)d_in[3];  const float* bn2 = (const float*)d_in[4];
    const float* W3  = (const float*)d_in[5];  const float* bn3 = (const float*)d_in[6];
    const float* W4  = (const float*)d_in[7];  const float* bn4 = (const float*)d_in[8];
    const float* W5  = (const float*)d_in[9];  const float* bn5 = (const float*)d_in[10];
    const float* Wh1 = (const float*)d_in[11]; const float* bh1 = (const float*)d_in[12];
    const float* bn6 = (const float*)d_in[13];
    const float* Wh2 = (const float*)d_in[14]; const float* bh2 = (const float*)d_in[15];
    const float* bn7 = (const float*)d_in[16];
    const float* Wh3 = (const float*)d_in[17]; const float* bh3 = (const float*)d_in[18];
    float* out = (float*)d_out;

    const size_t BN = (size_t)BATCH * NPTS;
    char* ws = (char*)d_ws;
    size_t off = 0;
    auto alloc = [&](size_t bytes) { size_t o = off; off = (off + bytes + 255) & ~(size_t)255; return o; };

    int*      idx  = (int*)     (ws + alloc(BN * KNN_K * sizeof(int)));
    float*    x1   = (float*)   (ws + alloc(BN * 64  * sizeof(float)));
    float*    x2   = (float*)   (ws + alloc(BN * 64  * sizeof(float)));
    float*    x3   = (float*)   (ws + alloc(BN * 128 * sizeof(float)));
    float*    x4   = (float*)   (ws + alloc(BN * 256 * sizeof(float)));
    _Float16* cat  = (_Float16*)(ws + alloc(BN * 512 * sizeof(_Float16)));
    float*    yb   = (float*)   (ws + alloc(BN * 512 * sizeof(float)));
    float*    g    = (float*)   (ws + alloc((size_t)BATCH * 1024 * sizeof(float)));
    float*    h1   = (float*)   (ws + alloc((size_t)BATCH * 512 * sizeof(float)));
    float*    h2   = (float*)   (ws + alloc((size_t)BATCH * 256 * sizeof(float)));
    _Float16* Wt1  = (_Float16*)(ws + alloc((size_t)32  * 64  * sizeof(_Float16)));
    _Float16* Wt2  = (_Float16*)(ws + alloc((size_t)128 * 64  * sizeof(_Float16)));
    _Float16* Wt3  = (_Float16*)(ws + alloc((size_t)128 * 128 * sizeof(_Float16)));
    _Float16* Wt4  = (_Float16*)(ws + alloc((size_t)256 * 256 * sizeof(_Float16)));
    _Float16* Wt5  = (_Float16*)(ws + alloc((size_t)512 * 512 * sizeof(_Float16)));

    // --- weight conversion: f32 [O][Cin] -> f16 padded [O][CP] ---
    convert_wt_kernel<<<(64 * 32   + 255) / 256, 256, 0, stream>>>(W1, Wt1, 64,  6,   32);
    convert_wt_kernel<<<(64 * 128  + 255) / 256, 256, 0, stream>>>(W2, Wt2, 64,  128, 128);
    convert_wt_kernel<<<(128 * 128 + 255) / 256, 256, 0, stream>>>(W3, Wt3, 128, 128, 128);
    convert_wt_kernel<<<(256 * 256 + 255) / 256, 256, 0, stream>>>(W4, Wt4, 256, 256, 256);
    convert_wt_kernel<<<(512 * 512 + 255) / 256, 256, 0, stream>>>(W5, Wt5, 512, 512, 512);

    const int nblk = (int)BN;   // 8192 (b,n) blocks

    // --- EdgeConv stack ---
    knn_kernel<3>  <<<nblk, 256, 0, stream>>>(x, idx);
    edgeconv_kernel<3, 64>   <<<nblk, 256, 0, stream>>>(x,  idx, Wt1, bn1, x1, cat, 0);
    knn_kernel<64> <<<nblk, 256, 0, stream>>>(x1, idx);
    edgeconv_kernel<64, 64>  <<<nblk, 256, 0, stream>>>(x1, idx, Wt2, bn2, x2, cat, 64);
    knn_kernel<64> <<<nblk, 256, 0, stream>>>(x2, idx);
    edgeconv_kernel<64, 128> <<<nblk, 256, 0, stream>>>(x2, idx, Wt3, bn3, x3, cat, 128);
    knn_kernel<128><<<nblk, 256, 0, stream>>>(x3, idx);
    edgeconv_kernel<128, 256><<<nblk, 256, 0, stream>>>(x3, idx, Wt4, bn4, x4, cat, 256);

    // --- W5 GEMM (8192 x 512 x 512) ---
    gemm_bn_kernel<<<(int)(BN / 16) * 32 / 8, 256, 0, stream>>>(cat, Wt5, bn5, yb);

    // --- pooling ---
    pool_kernel<<<BATCH * 512 / 8, 256, 0, stream>>>(yb, g);

    // --- MLP head ---
    fc_kernel<true,  true ><<<(BATCH * 512 + 255) / 256, 256, 0, stream>>>(g,  Wh1, bh1, bn6, h1, BATCH, 1024, 512);
    fc_kernel<true,  true ><<<(BATCH * 256 + 255) / 256, 256, 0, stream>>>(h1, Wh2, bh2, bn7, h2, BATCH, 512, 256);
    fc_kernel<false, false><<<(BATCH * 40  + 255) / 256, 256, 0, stream>>>(h2, Wh3, bh3, nullptr, out, BATCH, 256, 40);

    (void)in_sizes; (void)n_in; (void)out_size; (void)ws_size;
}
